// LSTM_Model_6476810682971
// MI455X (gfx1250) — matrix operand order repaired and checked by reading, not compiled, not run on hardware
//
#include <hip/hip_runtime.h>
#include <hip/hip_bf16.h>
#include <math.h>

typedef __attribute__((ext_vector_type(16))) _Float16 v16h;
typedef __attribute__((ext_vector_type(8)))  _Float16 v8h;
typedef __attribute__((ext_vector_type(4)))  _Float16 v4h;
typedef __attribute__((ext_vector_type(8)))  float    v8f;

#define BATCH   32768
#define TSTEPS  50
#define HDIM    128
#define CTXD    32
#define NGATES  512      // 4*H
#define MTILE   32       // batch rows per workgroup
#define THREADS 512      // 16 waves (wave32)
#define KPH     136      // LDS row pitch in halves (K max = 128, padded for banks)

__device__ __forceinline__ float sigf(float x){ return 1.0f/(1.0f + __expf(-x)); }

// A-matrix 16x32 fp16 fragment (wave32 layout per CDNA5 ISA):
// lane<16: row=lane,     a[0..7]=K k0..k0+7,  a[8..15]=K k0+16..k0+23
// lane>=16: row=lane-16, a[0..7]=K k0+8..15,  a[8..15]=K k0+24..31
__device__ __forceinline__ v16h load_a_frag(const _Float16* __restrict__ base,
                                            int lane, int k0){
  const int half = lane >> 4;
  const int row  = lane & 15;
  const _Float16* p = base + row*KPH + k0 + half*8;
  v8h lo = *(const v8h*)(p);
  v8h hi = *(const v8h*)(p + 16);
  v16h a;
  #pragma unroll
  for (int i=0;i<8;++i){ a[i]=lo[i]; a[i+8]=hi[i]; }
  return a;
}

// B-matrix 32x16 fp16 fragment: lane<16: col N=lane, b[0..7]=K k0..k0+7, b[8..15]=K k0+16..k0+23;
// lane>=16: col N=lane-16, b[0..7]=K k0+8..15, b[8..15]=K k0+24..31. W stored row-major [n][k].
__device__ __forceinline__ v16h load_b_frag(const _Float16* __restrict__ w,
                                            int tile, int lane, int k0){
  const int n = tile*16 + (lane & 15);
  const _Float16* p = w + (size_t)n*KPH + k0 + ((lane >> 4) << 3);
  v8h b0 = *(const v8h*)(p);
  v8h b1 = *(const v8h*)(p + 16);
  v16h b;
  #pragma unroll
  for (int i=0;i<8;++i){ b[i]=b0[i]; b[i+8]=b1[i]; }
  return b;
}

__global__ __launch_bounds__(THREADS)
void lstm_wmma_kernel(const float* __restrict__ X_info,
                      const int*   __restrict__ X_lob,
                      const int*   __restrict__ X_cc,
                      const int*   __restrict__ X_inj,
                      const float* __restrict__ X_seq,
                      const float* __restrict__ E_lob,
                      const float* __restrict__ E_cc,
                      const float* __restrict__ E_inj,
                      const float* __restrict__ W_ctx,
                      const float* __restrict__ b_ctx,
                      const float* __restrict__ W_ih,
                      const float* __restrict__ W_hh,
                      const float* __restrict__ b_ih,
                      const float* __restrict__ b_hh,
                      const float* __restrict__ W_ind,
                      const float* __restrict__ b_ind,
                      const float* __restrict__ W_pay,
                      const float* __restrict__ b_pay,
                      float* __restrict__ out)
{
  __shared__ __align__(16) _Float16 Wlds[NGATES*KPH];   // W_ih (init) then W_hh
  __shared__ __align__(16) _Float16 Abuf[2*MTILE*KPH];  // ping-pong h (fp16)
  __shared__ float s_x[MTILE][4];

  const int tid  = threadIdx.x;
  const int lane = tid & 31;
  const int wave = tid >> 5;
  const int m    = wave >> 3;          // batch M-tile (0/1)
  const int j    = wave & 7;           // 16-col block within each gate
  const int bg0  = blockIdx.x * MTILE;
  const int hrow = tid >> 4;           // heads: row (0..31)
  const int hl   = tid & 15;           // heads: lane within row group

  // ---- stage W_ih (fp16, K padded to 64) into Wlds; clear Abuf[0] ----
  {
    const float4* src = (const float4*)(W_ih + tid*36);   // 144B rows, 16B aligned
    _Float16* wr = Wlds + (size_t)tid*KPH;
    #pragma unroll
    for (int q=0;q<4;++q){
      float4 a = src[2*q], b = src[2*q+1];
      v8h h;
      h[0]=(_Float16)a.x; h[1]=(_Float16)a.y; h[2]=(_Float16)a.z; h[3]=(_Float16)a.w;
      h[4]=(_Float16)b.x; h[5]=(_Float16)b.y; h[6]=(_Float16)b.z; h[7]=(_Float16)b.w;
      *(v8h*)(wr + q*8) = h;
    }
    {
      float4 a = src[8];
      v4h h; h[0]=(_Float16)a.x; h[1]=(_Float16)a.y; h[2]=(_Float16)a.z; h[3]=(_Float16)a.w;
      *(v4h*)(wr + 32) = h;
    }
    v8h z = (v8h)(_Float16)0.0f;
    *(v4h*)(wr + 36) = *(const v4h*)&z;
    #pragma unroll
    for (int q=5;q<8;++q) *(v8h*)(wr + q*8) = z;          // cols 40..64 = 0
  }
  if (tid < MTILE){
    _Float16* ar = Abuf + (size_t)tid*KPH;
    v8h z = (v8h)(_Float16)0.0f;
    #pragma unroll
    for (int q=0;q<17;++q) *(v8h*)(ar + q*8) = z;         // KPH=136 halves
  }
  // per-lane constant registers
  float wih4[4][4];   // W_ih[n][0..3], n = nt*128 + j*16 + (lane&15)
  float biasv[4];
  #pragma unroll
  for (int nt=0; nt<4; ++nt){
    int n = nt*128 + j*16 + (lane & 15);
    #pragma unroll
    for (int c=0;c<4;++c) wih4[nt][c] = W_ih[n*36 + c];
    biasv[nt] = b_ih[n] + b_hh[n];
  }
  float wind[8], wpay[8];
  {
    int ks = hl * 8;
    #pragma unroll
    for (int i=0;i<8;++i){ wind[i]=W_ind[ks+i]; wpay[i]=W_pay[ks+i]; }
  }
  const float bind0 = b_ind[0];
  const float bpay0 = b_pay[0];
  // s_x for t=0 (reference guarantees no NaN at t=0; prev preds are 0 anyway)
  if (hl < 4){
    float v = X_seq[((size_t)(bg0+hrow)*TSTEPS)*4 + hl];
    if (hl < 2 && __builtin_isnan(v)) v = 0.0f;
    s_x[hrow][hl] = v;
  }
  __syncthreads();

  // ---- ctx = [X_info | E_lob | E_cc | E_inj] @ W_ctx.T + b_ctx -> Abuf[0][.,4..36]
  for (int p = tid; p < MTILE*CTXD; p += THREADS){
    int row = p >> 5, oc = p & 31;
    int bg = bg0 + row;
    const float* wr = W_ctx + oc*43;
    float s = b_ctx[oc];
    #pragma unroll
    for (int k=0;k<3;++k)  s += X_info[bg*3+k] * wr[k];
    int il = X_lob[bg];
    #pragma unroll
    for (int k=0;k<8;++k)  s += E_lob[il*8+k]  * wr[3+k];
    int ic = X_cc[bg];
    #pragma unroll
    for (int k=0;k<16;++k) s += E_cc[ic*16+k]  * wr[11+k];
    int ii = X_inj[bg];
    #pragma unroll
    for (int k=0;k<16;++k) s += E_inj[ii*16+k] * wr[27+k];
    Abuf[row*KPH + 4 + oc] = (_Float16)s;
  }
  __syncthreads();

  // ---- gctx = [0(x),ctx,pad] @ W_ih.T + bias (constant over time) ----
  v8f gctx[4];
  #pragma unroll
  for (int nt=0;nt<4;++nt){
    #pragma unroll
    for (int r=0;r<8;++r) gctx[nt][r] = biasv[nt];
  }
  #pragma unroll
  for (int k=0;k<2;++k){
    v16h a = load_a_frag(Abuf + m*16*KPH, lane, k*32);
    #pragma unroll
    for (int nt=0;nt<4;++nt){
      v16h b = load_b_frag(Wlds, j + 8*nt, lane, k*32);
      gctx[nt] = __builtin_amdgcn_wmma_f32_16x16x32_f16(false, a, false, b,
                                                        (short)0, gctx[nt],
                                                        false, false);
    }
  }
  __syncthreads();

  // ---- stage W_hh (fp16) into Wlds; clear Abuf[0] ctx region (h := 0) ----
  {
    const float4* src = (const float4*)(W_hh + tid*128);
    _Float16* wr = Wlds + (size_t)tid*KPH;
    #pragma unroll
    for (int q=0;q<16;++q){
      float4 a = src[2*q], b = src[2*q+1];
      v8h h;
      h[0]=(_Float16)a.x; h[1]=(_Float16)a.y; h[2]=(_Float16)a.z; h[3]=(_Float16)a.w;
      h[4]=(_Float16)b.x; h[5]=(_Float16)b.y; h[6]=(_Float16)b.z; h[7]=(_Float16)b.w;
      *(v8h*)(wr + q*8) = h;
    }
  }
  if (tid < MTILE){
    _Float16* ar = Abuf + (size_t)tid*KPH;
    v8h z = (v8h)(_Float16)0.0f;
    #pragma unroll
    for (int q=0;q<8;++q) *(v8h*)(ar + q*8) = z;          // cols 0..64 = 0
  }
  __syncthreads();

  v8f cfrag;                              // persistent cell state [16x16] frag
  #pragma unroll
  for (int r=0;r<8;++r) cfrag[r] = 0.0f;

  const int rbase = m*16 + 8*(lane >> 4); // frag row base (C/D layout)

  for (int t=0; t<TSTEPS; ++t){
    const _Float16* Acur = Abuf + (size_t)(t & 1)*MTILE*KPH;
    _Float16*       Anxt = Abuf + (size_t)((t+1) & 1)*MTILE*KPH;

    // prefetch next-step x early; latency hides behind the matmul
    float xn = 0.0f;
    if (hl < 4 && t+1 < TSTEPS)
      xn = X_seq[((size_t)(bg0+hrow)*TSTEPS + (t+1))*4 + hl];

    // gates = gctx + x[0..3]*W_ih[:,0:4] + h @ W_hh.T
    v8f acc[4];
    #pragma unroll
    for (int r=0;r<8;++r){
      const float x0 = s_x[rbase+r][0], x1 = s_x[rbase+r][1];
      const float x2 = s_x[rbase+r][2], x3 = s_x[rbase+r][3];
      #pragma unroll
      for (int nt=0;nt<4;++nt){
        acc[nt][r] = gctx[nt][r] + x0*wih4[nt][0] + x1*wih4[nt][1]
                                 + x2*wih4[nt][2] + x3*wih4[nt][3];
      }
    }
    #pragma unroll
    for (int k=0;k<4;++k){
      v16h a = load_a_frag(Acur + m*16*KPH, lane, k*32);
      #pragma unroll
      for (int nt=0;nt<4;++nt){
        v16h b = load_b_frag(Wlds, j + 8*nt, lane, k*32);
        acc[nt] = __builtin_amdgcn_wmma_f32_16x16x32_f16(false, a, false, b,
                                                         (short)0, acc[nt],
                                                         false, false);
      }
    }

    // LSTM cell: gates i,f,g,o = acc[0..3]
    float hreg[8];
    #pragma unroll
    for (int r=0;r<8;++r){
      float iv = sigf(acc[0][r]);
      float fv = sigf(acc[1][r]);
      float gv = tanhf(acc[2][r]);
      float ov = sigf(acc[3][r]);
      float cn = fv*cfrag[r] + iv*gv;
      cfrag[r] = cn;
      hreg[r]  = ov * tanhf(cn);
    }

    // h_new -> fp16 -> other buffer (no read/write hazard: readers used Acur)
    {
      _Float16* hp = Anxt + (size_t)rbase*KPH + j*16 + (lane & 15);
      #pragma unroll
      for (int r=0;r<8;++r) hp[r*KPH] = (_Float16)hreg[r];
    }
    __syncthreads();

    // heads + fused autoregressive x-build for t+1
    {
      const _Float16* hp = Anxt + (size_t)hrow*KPH + hl*8;
      v8h hv = *(const v8h*)hp;
      float si = 0.f, sp = 0.f;
      #pragma unroll
      for (int i=0;i<8;++i){
        float h = (float)hv[i];
        si += h*wind[i]; sp += h*wpay[i];
      }
      #pragma unroll
      for (int off=8; off>0; off>>=1){
        si += __shfl_xor(si, off, 16);
        sp += __shfl_xor(sp, off, 16);
      }
      if (hl < 4){
        float ind = sigf(si + bind0);
        float pay = sp + bpay0;
        if (hl == 0){
          size_t bg = (size_t)(bg0 + hrow);
          out[bg*TSTEPS + t] = ind;
          out[(size_t)BATCH*TSTEPS + bg*TSTEPS + t] = pay;
        }
        float v = xn;
        if (hl < 2 && __builtin_isnan(v)) v = (hl == 0) ? ind : ind*pay;
        s_x[hrow][hl] = v;
      }
    }
    __syncthreads();
  }
}

extern "C" void kernel_launch(void* const* d_in, const int* in_sizes, int n_in,
                              void* d_out, int out_size, void* d_ws, size_t ws_size,
                              hipStream_t stream) {
  const float* X_info = (const float*)d_in[0];
  const int*   X_lob  = (const int*)  d_in[1];
  const int*   X_cc   = (const int*)  d_in[2];
  const int*   X_inj  = (const int*)  d_in[3];
  const float* X_seq  = (const float*)d_in[4];
  const float* E_lob  = (const float*)d_in[5];
  const float* E_cc   = (const float*)d_in[6];
  const float* E_inj  = (const float*)d_in[7];
  const float* W_ctx  = (const float*)d_in[8];
  const float* b_ctx  = (const float*)d_in[9];
  const float* W_ih   = (const float*)d_in[10];
  const float* W_hh   = (const float*)d_in[11];
  const float* b_ih   = (const float*)d_in[12];
  const float* b_hh   = (const float*)d_in[13];
  const float* W_ind  = (const float*)d_in[14];
  const float* b_ind  = (const float*)d_in[15];
  const float* W_pay  = (const float*)d_in[16];
  const float* b_pay  = (const float*)d_in[17];
  float* out = (float*)d_out;

  hipLaunchKernelGGL(lstm_wmma_kernel, dim3(BATCH/MTILE), dim3(THREADS), 0, stream,
                     X_info, X_lob, X_cc, X_inj, X_seq, E_lob, E_cc, E_inj,
                     W_ctx, b_ctx, W_ih, W_hh, b_ih, b_hh,
                     W_ind, b_ind, W_pay, b_pay, out);
}
